// SingleHeadAttention_40638980555312
// MI455X (gfx1250) — matrix operand order repaired and checked by reading, not compiled, not run on hardware
//
#include <hip/hip_runtime.h>
#include <hip/hip_bf16.h>

#define HD    128
#define DIM   2048
#define SEQ   4096
#define BATCH 4
#define MTOT  (BATCH * SEQ)   // 16384

typedef __attribute__((ext_vector_type(16))) __bf16 v16bf;
typedef __attribute__((ext_vector_type(8)))  float  v8f;

union AF { v16bf v; unsigned u[8]; };

__device__ __forceinline__ unsigned short f2bf(float f) {
  unsigned u = __float_as_uint(f);
  u += 0x7fffu + ((u >> 16) & 1u);          // round-to-nearest-even
  return (unsigned short)(u >> 16);
}

__device__ __forceinline__ v8f wmma_bf16(const AF& a, const AF& b, v8f c) {
  return __builtin_amdgcn_wmma_f32_16x16x32_bf16(
      /*neg_a=*/false, a.v, /*neg_b=*/false, b.v,
      /*c_mod=*/(short)0, c, /*reuse_a=*/false, /*reuse_b=*/false);
}

// LDS offset of a __shared__ object (generic-pointer low 32 bits == LDS addr)
__device__ __forceinline__ unsigned lds_off(const void* p) {
  return (unsigned)(size_t)p;
}

// Async global->LDS 16-byte copy (GVS mode: saddr base + 32-bit voffset).
__device__ __forceinline__ void async_copy16(unsigned lds_addr, unsigned goff,
                                             const void* base) {
  asm volatile("global_load_async_to_lds_b128 %0, %1, %2"
               :: "v"(lds_addr), "v"(goff), "s"(base) : "memory");
}

// -------------------------------------------------------------------------
// Kernel 0: convert W(q|k|v) fp32 [k][n] -> bf16 transposed WT[o][n][k].
// Tiny (3 MB read), runs once; lets the GEMM stage W tiles with pure
// async copies (no per-element transpose in the hot loop).
// -------------------------------------------------------------------------
__global__ __launch_bounds__(256, 1)
void wprep_kernel(const float* __restrict__ Wq, const float* __restrict__ Wk,
                  const float* __restrict__ Wv, unsigned short* __restrict__ WT)
{
  int idx = blockIdx.x * 256 + threadIdx.x;      // 0 .. 3*128*2048-1
  int o   = idx / (HD * DIM);
  int rem = idx % (HD * DIM);
  int n   = rem / DIM;
  int kk  = rem % DIM;
  const float* W = (o == 0) ? Wq : (o == 1) ? Wk : Wv;
  WT[idx] = f2bf(W[(size_t)kk * HD + n]);
}

// -------------------------------------------------------------------------
// Kernel 1: fused QKV projection — x read ONCE.
// Block = 256 threads (8 waves) computes 64 rows x (3 outputs x 128 cols).
// Wave -> (row-group = wave>>1 [16 rows], col-half = wave&1 [64 cols]).
// x tile converted f32->bf16 into LDS; W tiles double-buffered via async.
//   q, k  -> bf16 row-major   [b*S + s][HD]
//   v     -> bf16 TRANSPOSED  [b][HD][S]
// -------------------------------------------------------------------------
__global__ __launch_bounds__(256, 1)
void qkv_proj_kernel(const float* __restrict__ x,
                     const float* __restrict__ bq, const float* __restrict__ bk,
                     const float* __restrict__ bv,
                     const unsigned short* __restrict__ WT,
                     unsigned short* __restrict__ qkv)
{
  __shared__ unsigned short lA[64 * 32];          // x tile [row][k] (bf16)
  __shared__ unsigned short lW[2][3][128 * 32];   // W tiles [o][n][k], dbl-buf

  const int tid   = threadIdx.x;
  const int wave  = tid >> 5;
  const int lane  = tid & 31;
  const int hi    = lane >> 4;
  const int ln    = lane & 15;
  const int rowg  = wave >> 1;        // 0..3  (16-row group)
  const int nhalf = wave & 1;         // 0..1  (64-col half)
  const int row0  = blockIdx.x * 64;

  v8f acc[3][4];
  #pragma unroll
  for (int o = 0; o < 3; ++o)
    #pragma unroll
    for (int t = 0; t < 4; ++t)
      #pragma unroll
      for (int j = 0; j < 8; ++j) acc[o][t][j] = 0.0f;

  const int kbA = hi * 16;
  const int kbB = hi * 16;

  // stage one 32-wide K-slab of all three W matrices (1536 x 16B chunks)
  auto issue_W = [&](int bufsel, int kk) {
    unsigned ldsW = lds_off(&lW[bufsel][0][0]);
    #pragma unroll
    for (int t = 0; t < 6; ++t) {
      int c = tid + t * 256;                      // 0..1535
      int o = c >> 9;                             // output 0..2
      int r = c & 511;                            // chunk in 8KB tile
      int n = r >> 2, part = r & 3;
      async_copy16(ldsW + c * 16,
                   (unsigned)(((size_t)o * HD * DIM + (size_t)n * DIM + kk) * 2)
                       + part * 16,
                   WT);
    }
  };

  issue_W(0, 0);

  for (int t = 0; t < DIM / 32; ++t) {
    const int kk  = t * 32;
    const int buf = t & 1;
    if (t + 1 < DIM / 32) issue_W(buf ^ 1, kk + 32);   // prefetch next slab

    // stage x tile (64x32 f32 -> bf16); lA free: prior readers passed
    // the end-of-step barrier.
    #pragma unroll
    for (int it = 0; it < 2; ++it) {
      int i  = tid + it * 256;                    // 512 float4 chunks
      int r  = i >> 3;
      int c4 = (i & 7) * 4;
      const float4 xv = *(const float4*)(&x[(size_t)(row0 + r) * DIM + kk + c4]);
      unsigned short* p = &lA[r * 32 + c4];
      p[0] = f2bf(xv.x); p[1] = f2bf(xv.y); p[2] = f2bf(xv.z); p[3] = f2bf(xv.w);
    }

    if (t + 1 < DIM / 32)
      asm volatile("s_wait_asynccnt 0x6" ::: "memory");  // current slab done
    else
      asm volatile("s_wait_asynccnt 0x0" ::: "memory");
    __syncthreads();

    // A fragment (16 rows x K=32), shared by all 12 WMMAs this step
    AF a;
    const unsigned short* ap = &lA[(rowg * 16 + ln) * 32];
    #pragma unroll
    for (int vv = 0; vv < 8; ++vv) {
      int kidx = vv * 2 + kbA;
      a.u[vv] = *(const unsigned*)(ap + kidx);
    }
    #pragma unroll
    for (int o = 0; o < 3; ++o)
      #pragma unroll
      for (int nt = 0; nt < 4; ++nt) {
        const int ntile = nhalf * 4 + nt;
        AF b;
        const unsigned short* bp = &lW[buf][o][(ntile * 16 + ln) * 32];
        #pragma unroll
        for (int vv = 0; vv < 8; ++vv) {
          int kidx = vv * 2 + kbB;
          b.u[vv] = *(const unsigned*)(bp + kidx);
        }
        acc[o][nt] = wmma_bf16(a, b, acc[o][nt]);
      }
    __syncthreads();
  }

  // epilogue: bias add + bf16 store.  D layout: VGPR i -> row i + 8*hi.
  #pragma unroll
  for (int o = 0; o < 3; ++o) {
    const float* bias = (o == 0) ? bq : (o == 1) ? bk : bv;
    if (o != 2) {
      unsigned short* dst = qkv + (size_t)o * MTOT * HD;
      #pragma unroll
      for (int nt = 0; nt < 4; ++nt) {
        const int col = nhalf * 64 + nt * 16 + ln;
        const float bn = bias[col];
        #pragma unroll
        for (int i = 0; i < 8; ++i) {
          int row = row0 + rowg * 16 + i + hi * 8;
          dst[(size_t)row * HD + col] = f2bf(acc[o][nt][i] + bn);
        }
      }
    } else {
      // V transposed: vT[b][h][s]; each lane packs 8 consecutive s into 16B
      unsigned short* vT = qkv + (size_t)2 * MTOT * HD;
      const int bi = row0 / SEQ;
      const int s0 = (row0 % SEQ) + rowg * 16 + hi * 8;
      #pragma unroll
      for (int nt = 0; nt < 4; ++nt) {
        const int h = nhalf * 64 + nt * 16 + ln;
        const float bn = bias[h];
        unsigned pk[4];
        #pragma unroll
        for (int i = 0; i < 4; ++i) {
          unsigned lo = f2bf(acc[o][nt][2 * i]     + bn);
          unsigned hv = f2bf(acc[o][nt][2 * i + 1] + bn);
          pk[i] = lo | (hv << 16);
        }
        uint4 val; val.x = pk[0]; val.y = pk[1]; val.z = pk[2]; val.w = pk[3];
        *(uint4*)&vT[((size_t)bi * HD + h) * SEQ + s0] = val;
      }
    }
  }
}

// -------------------------------------------------------------------------
// Kernel 2: causal flash attention, one batch per blockIdx.y.
// Block = 256 threads (8 waves); each wave owns 16 q rows; 128 rows/block.
// 64-key blocks, K/V double-buffered via global_load_async_to_lds_b128.
// Online softmax; post-softmax 1/sqrt(H) fused into epilogue.
// -------------------------------------------------------------------------
__global__ __launch_bounds__(256, 1)
void attn_kernel(const unsigned short* __restrict__ q,
                 const unsigned short* __restrict__ k,
                 const unsigned short* __restrict__ vT,
                 float* __restrict__ out)
{
  __shared__ unsigned short lK[2][64 * HD];   // key tile   [j][h]     16KB x2
  __shared__ unsigned short lV[2][HD * 64];   // value tile [h][j]     16KB x2
  __shared__ unsigned short lP[8][16 * 64];   // per-wave P [row][j]   16KB

  const int tid  = threadIdx.x;
  const int wave = tid >> 5;
  const int lane = tid & 31;
  const int hi   = lane >> 4;
  const int ln   = lane & 15;
  const int b    = blockIdx.y;
  const int qr0  = blockIdx.x * 128;

  const unsigned short* qb  = q  + (size_t)b * SEQ * HD;
  const unsigned short* kbp = k  + (size_t)b * SEQ * HD;
  const unsigned short* vbp = vT + (size_t)b * HD * SEQ;
  float* ob = out + (size_t)b * SEQ * HD;

  const int kbA = hi * 16;
  const int kbB = hi * 16;
  const int rowbase = qr0 + wave * 16;

  // Q fragments for full head dim (4 K-steps of 32), register-resident
  AF qf[4];
  {
    const unsigned short* qp = qb + (size_t)(rowbase + ln) * HD;
    #pragma unroll
    for (int s = 0; s < 4; ++s)
      #pragma unroll
      for (int vv = 0; vv < 8; ++vv) {
        int h = s * 32 + vv * 2 + kbA;
        qf[s].u[vv] = *(const unsigned*)(qp + h);
      }
  }

  v8f o[8];
  #pragma unroll
  for (int t = 0; t < 8; ++t)
    #pragma unroll
    for (int j = 0; j < 8; ++j) o[t][j] = 0.0f;

  float mrow[8], lrow[8];
  #pragma unroll
  for (int i = 0; i < 8; ++i) { mrow[i] = -__builtin_inff(); lrow[i] = 0.0f; }

  const int nkb = qr0 / 64 + 2;   // 64-key blocks up to key qr0+127

  // async-stage one 64-key K/V tile pair (1024 x 16B chunks each)
  auto issue_tile = [&](int bufsel, int key0) {
    unsigned ldsK = lds_off(&lK[bufsel][0]);
    unsigned ldsV = lds_off(&lV[bufsel][0]);
    #pragma unroll
    for (int t = 0; t < 4; ++t) {
      int c = tid + t * 256;                        // 0..1023
      // K tile: row-major contiguous copy (16KB)
      async_copy16(ldsK + c * 16,
                   (unsigned)((size_t)key0 * HD * 2) + c * 16, kbp);
      // V tile: 128 rows of vT, 128B per row (8 chunks)
      int h = c >> 3, part = c & 7;
      async_copy16(ldsV + c * 16,
                   (unsigned)(((size_t)h * SEQ + key0) * 2) + part * 16, vbp);
    }
  };

  issue_tile(0, 0);

  for (int kblk = 0; kblk < nkb; ++kblk) {
    const int key0 = kblk * 64;
    const int cur  = kblk & 1;

    if (kblk + 1 < nkb) {
      issue_tile(cur ^ 1, key0 + 64);               // prefetch next tile
      asm volatile("s_wait_asynccnt 0x8" ::: "memory");  // current tile done
    } else {
      asm volatile("s_wait_asynccnt 0x0" ::: "memory");
    }
    __syncthreads();

    if (key0 <= rowbase + 15) {   // skip fully-masked diagonal-tail blocks
      // S = Q K^T  (16 rows x 64 keys)
      v8f sacc[4];
      #pragma unroll
      for (int t = 0; t < 4; ++t)
        #pragma unroll
        for (int j = 0; j < 8; ++j) sacc[t][j] = 0.0f;

      #pragma unroll
      for (int s = 0; s < 4; ++s)
        #pragma unroll
        for (int nt = 0; nt < 4; ++nt) {
          AF bfr;
          const unsigned short* bp = &lK[cur][(nt * 16 + ln) * HD + s * 32];
          #pragma unroll
          for (int vv = 0; vv < 8; ++vv) {
            int h = vv * 2 + kbB;
            bfr.u[vv] = *(const unsigned*)(bp + h);
          }
          sacc[nt] = wmma_bf16(qf[s], bfr, sacc[nt]);
        }

      // causal mask near the diagonal
      if (key0 + 63 > rowbase) {
        #pragma unroll
        for (int nt = 0; nt < 4; ++nt) {
          int kg = key0 + nt * 16 + ln;
          #pragma unroll
          for (int i = 0; i < 8; ++i) {
            int rg = rowbase + i + hi * 8;
            if (kg > rg) sacc[nt][i] = -__builtin_inff();
          }
        }
      }

      // online softmax (one butterfly per row; VGPR i holds row i + 8*hi)
      unsigned short* pw = lP[wave];
      #pragma unroll
      for (int i = 0; i < 8; ++i) {
        float t = fmaxf(fmaxf(sacc[0][i], sacc[1][i]),
                        fmaxf(sacc[2][i], sacc[3][i]));
        #pragma unroll
        for (int off = 1; off < 16; off <<= 1)
          t = fmaxf(t, __shfl_xor(t, off, 32));
        float mnew  = fmaxf(mrow[i], t);
        float scale = __expf(mrow[i] - mnew);
        float p0 = __expf(sacc[0][i] - mnew);
        float p1 = __expf(sacc[1][i] - mnew);
        float p2 = __expf(sacc[2][i] - mnew);
        float p3 = __expf(sacc[3][i] - mnew);
        float ls = (p0 + p1) + (p2 + p3);
        #pragma unroll
        for (int off = 1; off < 16; off <<= 1)
          ls += __shfl_xor(ls, off, 32);
        lrow[i] = lrow[i] * scale + ls;
        mrow[i] = mnew;
        // restage P row (D layout -> A layout) while stats reduce
        const int prow = (i + hi * 8) * 64 + ln;
        pw[prow +  0] = f2bf(p0);
        pw[prow + 16] = f2bf(p1);
        pw[prow + 32] = f2bf(p2);
        pw[prow + 48] = f2bf(p3);
        #pragma unroll
        for (int t8 = 0; t8 < 8; ++t8) o[t8][i] *= scale;
      }

      asm volatile("s_wait_dscnt 0x0" ::: "memory");   // intra-wave LDS RAW

      // O += P @ V  (two K=32 steps over the 64 keys)
      #pragma unroll
      for (int ks = 0; ks < 2; ++ks) {
        AF pf;
        const unsigned short* pp = &pw[ln * 64 + ks * 32];
        #pragma unroll
        for (int vv = 0; vv < 8; ++vv) {
          int kidx = vv * 2 + kbA;
          pf.u[vv] = *(const unsigned*)(pp + kidx);
        }
        #pragma unroll
        for (int nt = 0; nt < 8; ++nt) {
          AF bfr;
          const unsigned short* bp = &lV[cur][(nt * 16 + ln) * 64 + ks * 32];
          #pragma unroll
          for (int vv = 0; vv < 8; ++vv) {
            int j = vv * 2 + kbB;
            bfr.u[vv] = *(const unsigned*)(bp + j);
          }
          o[nt] = wmma_bf16(pf, bfr, o[nt]);
        }
      }
    }
    __syncthreads();
  }

  // epilogue: out = O / l / sqrt(H)   (reference scales AFTER softmax)
  const float rsH = 0.088388347648318447f;   // 1/sqrt(128)
  float fin[8];
  #pragma unroll
  for (int i = 0; i < 8; ++i) fin[i] = rsH / lrow[i];
  #pragma unroll
  for (int nt = 0; nt < 8; ++nt)
    #pragma unroll
    for (int i = 0; i < 8; ++i) {
      int row = rowbase + i + hi * 8;
      ob[(size_t)row * HD + nt * 16 + ln] = o[nt][i] * fin[i];
    }
}

// -------------------------------------------------------------------------
extern "C" void kernel_launch(void* const* d_in, const int* in_sizes, int n_in,
                              void* d_out, int out_size, void* d_ws, size_t ws_size,
                              hipStream_t stream)
{
  const float* x  = (const float*)d_in[0];
  const float* Wq = (const float*)d_in[1];
  const float* bq = (const float*)d_in[2];
  const float* Wk = (const float*)d_in[3];
  const float* bk = (const float*)d_in[4];
  const float* Wv = (const float*)d_in[5];
  const float* bv = (const float*)d_in[6];
  float* out = (float*)d_out;

  unsigned short* qkv = (unsigned short*)d_ws;            // bf16 q | k | vT
  unsigned short* WT  = qkv + (size_t)3 * MTOT * HD;      // bf16 WT[3][n][k]

  wprep_kernel<<<(3 * HD * DIM) / 256, 256, 0, stream>>>(Wq, Wk, Wv, WT);

  qkv_proj_kernel<<<MTOT / 64, 256, 0, stream>>>(x, bq, bk, bv, WT, qkv);

  const unsigned short* qp = qkv;
  const unsigned short* kp = qkv + (size_t)MTOT * HD;
  const unsigned short* vp = qkv + (size_t)2 * MTOT * HD;
  attn_kernel<<<dim3(SEQ / 128, BATCH), 256, 0, stream>>>(qp, kp, vp, out);
}
